// MultiCA_11115375362890
// MI455X (gfx1250) — compile-verified, matrix operand order checked by reading, not run on hardware
//
#include <hip/hip_runtime.h>

// ---------------------------------------------------------------------------
// Cross-attention B=2, T=2048, C=1024, H=16, D=64 on gfx1250 (CDNA5, wave32).
// Pipeline: (0) f32->bf16 conversion pass, (1) fused Q/K/V projections,
// (2) flash attention, (3) output projection. All matmuls are
// v_wmma_f32_16x16x32_bf16; all hot-loop tile loads use async global->LDS
// copies (ASYNCcnt) with double-buffered LDS so WMMA overlaps the DMA.
// ---------------------------------------------------------------------------

typedef __attribute__((ext_vector_type(16))) __bf16 v16bf;
typedef __attribute__((ext_vector_type(8)))  float  v8f;
typedef __attribute__((ext_vector_type(4)))  int    v4i;

#define LOG2E 1.44269504088896340736f

#if defined(__AMDGCN__) && __has_builtin(__builtin_amdgcn_global_load_async_to_lds_b128)
#define ASYNC_CP 1
#else
#define ASYNC_CP 0
#endif

static __device__ inline v8f wmma_bf16(v16bf a, v16bf b, v8f c) {
  // 8 args: (neg_a, A, neg_b, B, c_mod, C, reuse_a, reuse_b)
  return __builtin_amdgcn_wmma_f32_16x16x32_bf16(false, a, false, b, (short)0, c,
                                                 false, false);
}

// Copy 16B global -> LDS. Async path bypasses VGPRs (GLOBAL_LOAD_ASYNC_TO_LDS_B128).
static __device__ inline void async_cp16(const __bf16* g, __bf16* l) {
#if ASYNC_CP
  __builtin_amdgcn_global_load_async_to_lds_b128(
      (__attribute__((address_space(1))) v4i*)(g),
      (__attribute__((address_space(3))) v4i*)(l), 0, 0);
#else
  *reinterpret_cast<float4*>(l) = *reinterpret_cast<const float4*>(g);
#endif
}

static __device__ inline void wait_async0() {
#if ASYNC_CP
#if __has_builtin(__builtin_amdgcn_s_wait_asynccnt)
  __builtin_amdgcn_s_wait_asynccnt(0);
#else
  asm volatile("s_wait_asynccnt 0x0" ::: "memory");
#endif
#endif
}

// 16x32 bf16 WMMA operand from row-major [16][stride] (A layout; B symmetric
// via rows of B^T). Lane l: row = l&15, K-chunks [half*8,+8) and [16+half*8,+8).
static __device__ inline v16bf load_frag(const __bf16* base, int stride) {
  const int lane = threadIdx.x & 31;
  const int row  = lane & 15;
  const int half = lane >> 4;
  const __bf16* p = base + row * stride + half * 8;
  v16bf r;
  reinterpret_cast<float4*>(&r)[0] = reinterpret_cast<const float4*>(p)[0];
  reinterpret_cast<float4*>(&r)[1] = reinterpret_cast<const float4*>(p + 16)[0];
  return r;
}

// ---------------------------------------------------------------------------
// Kernel 0: f32 -> bf16 conversion (one pass; everything stays L2-resident).
// ---------------------------------------------------------------------------
__global__ __launch_bounds__(256) void cvt_bf16(const float* __restrict__ s,
                                                __bf16* __restrict__ d, int n) {
  const int i = (blockIdx.x * 256 + threadIdx.x) * 8;
  if (i >= n) return;
  const float4 f0 = reinterpret_cast<const float4*>(s + i)[0];
  const float4 f1 = reinterpret_cast<const float4*>(s + i)[1];
  union { __bf16 h[8]; float4 q; } u;
  u.h[0] = (__bf16)f0.x; u.h[1] = (__bf16)f0.y;
  u.h[2] = (__bf16)f0.z; u.h[3] = (__bf16)f0.w;
  u.h[4] = (__bf16)f1.x; u.h[5] = (__bf16)f1.y;
  u.h[6] = (__bf16)f1.z; u.h[7] = (__bf16)f1.w;
  reinterpret_cast<float4*>(d + i)[0] = u.q;
}

// ---------------------------------------------------------------------------
// Kernel 1: fused QKV projections (all-bf16). blockIdx.z selects:
//   0: Q = Y@Wq^T -> [b,h,t,d]   1: K = Y@Wk^T -> [b,h,t,d]
//   2: V^T = X@Wv^T -> [b,h,d,t] (transposed so P@V B-operand is contiguous)
// Block tile 256(M)x128(N), 8 waves in 4x2 grid, wave tile 64x64 (16 WMMA/kstep).
// Double-buffered LDS fed by async copies.
// ---------------------------------------------------------------------------
__global__ __launch_bounds__(256) void qkv_proj(
    const __bf16* __restrict__ Ybf, const __bf16* __restrict__ Xbf,
    const __bf16* __restrict__ Wqbf, const __bf16* __restrict__ Wkbf,
    const __bf16* __restrict__ Wvbf,
    __bf16* __restrict__ Qb, __bf16* __restrict__ Kb, __bf16* __restrict__ Vtb) {
  constexpr int Kdim = 1024, T = 2048, H = 16, Dh = 64;
  __shared__ __attribute__((aligned(16))) __bf16 As[2][256][40];
  __shared__ __attribute__((aligned(16))) __bf16 Bs[2][128][40];

  const int tid  = threadIdx.x;
  const int mode = blockIdx.z;
  const __bf16* act = (mode == 2) ? Xbf : Ybf;
  const __bf16* W   = (mode == 0) ? Wqbf : (mode == 1) ? Wkbf : Wvbf;
  const long aBase = (long)blockIdx.y * 256 * Kdim;
  const long bBase = (long)blockIdx.x * 128 * Kdim;

  const int wv = tid >> 5, waveM = wv >> 1, waveN = wv & 1;
  const int lane = tid & 31, half = lane >> 4, ln = lane & 15;

  auto issueAB = [&](int buf, int kt) {
    // A: 256 rows x 32 k, one row per thread (4 x b128)
    const __bf16* gA = act + aBase + (long)tid * Kdim + kt;
    __bf16* lA = &As[buf][tid][0];
#pragma unroll
    for (int c = 0; c < 4; ++c) async_cp16(gA + c * 8, lA + c * 8);
    // B: 128 rows x 32 k, half row per thread (2 x b128)
    const int br = tid >> 1, bc = (tid & 1) * 16;
    const __bf16* gB = W + bBase + (long)br * Kdim + kt + bc;
    __bf16* lB = &Bs[buf][br][bc];
    async_cp16(gB, lB);
    async_cp16(gB + 8, lB + 8);
  };

  const v8f zero = {0.f, 0.f, 0.f, 0.f, 0.f, 0.f, 0.f, 0.f};
  v8f acc[4][4];
#pragma unroll
  for (int mi = 0; mi < 4; ++mi)
#pragma unroll
    for (int ni = 0; ni < 4; ++ni) acc[mi][ni] = zero;

  issueAB(0, 0);
  for (int kt = 0; kt < Kdim; kt += 32) {
    const int cur = (kt >> 5) & 1;
    wait_async0();
    __syncthreads();
    if (kt + 32 < Kdim) issueAB(cur ^ 1, kt + 32);

    v16bf a[4], b[4];
#pragma unroll
    for (int mi = 0; mi < 4; ++mi)
      a[mi] = load_frag(&As[cur][waveM * 64 + mi * 16][0], 40);
#pragma unroll
    for (int ni = 0; ni < 4; ++ni)
      b[ni] = load_frag(&Bs[cur][waveN * 64 + ni * 16][0], 40);
#pragma unroll
    for (int mi = 0; mi < 4; ++mi)
#pragma unroll
      for (int ni = 0; ni < 4; ++ni)
        acc[mi][ni] = wmma_bf16(a[mi], b[ni], acc[mi][ni]);
  }

#pragma unroll
  for (int mi = 0; mi < 4; ++mi)
#pragma unroll
    for (int ni = 0; ni < 4; ++ni)
#pragma unroll
      for (int r = 0; r < 8; ++r) {
        const int m = blockIdx.y * 256 + waveM * 64 + mi * 16 + r + 8 * half;
        const int n = blockIdx.x * 128 + waveN * 64 + ni * 16 + ln;
        const int bb = m >> 11, t = m & (T - 1);
        const int h = n >> 6, d = n & (Dh - 1);
        const __bf16 hv = (__bf16)acc[mi][ni][r];
        if (mode == 0)
          Qb[(((long)bb * H + h) * T + t) * Dh + d] = hv;
        else if (mode == 1)
          Kb[(((long)bb * H + h) * T + t) * Dh + d] = hv;
        else
          Vtb[(((long)bb * H + h) * Dh + d) * T + t] = hv;
      }
}

// ---------------------------------------------------------------------------
// Kernel 2: flash attention per (b, h, 128-query tile). 8 waves, each owns 16
// query rows over the full 128-key tile. Row stats via half-wave __shfl_xor.
// K/V tiles double-buffered in LDS via async copies.
// ---------------------------------------------------------------------------
__global__ __launch_bounds__(256) void attn(
    const __bf16* __restrict__ Qb, const __bf16* __restrict__ Kb,
    const __bf16* __restrict__ Vtb, __bf16* __restrict__ AO) {
  constexpr int T = 2048, H = 16, Dh = 64;
  __shared__ __attribute__((aligned(16))) __bf16 Ks[2][128][72];   // keys x d
  __shared__ __attribute__((aligned(16))) __bf16 Vs[2][64][136];   // d x keys
  __shared__ __attribute__((aligned(16))) __bf16 Ps[8][16][136];   // per-wave P

  const int tid = threadIdx.x;
  const int qtile = blockIdx.x, h = blockIdx.y, b = blockIdx.z;
  const int bh = b * H + h;
  const int wv = tid >> 5, lane = tid & 31, half = lane >> 4, ln = lane & 15;
  const int qRow0 = qtile * 128 + wv * 16;

  auto issueKV = [&](int buf, int kb) {
    {  // K tile: 128 keys x 64 d
      const int row = tid >> 1, ch = (tid & 1) * 32;
      const __bf16* g = Kb + ((long)bh * T + kb + row) * Dh + ch;
      __bf16* l = &Ks[buf][row][ch];
#pragma unroll
      for (int c = 0; c < 4; ++c) async_cp16(g + c * 8, l + c * 8);
    }
    {  // V^T tile: 64 d x 128 keys
      const int row = tid >> 2, ch = (tid & 3) * 32;
      const __bf16* g = Vtb + ((long)bh * Dh + row) * T + kb + ch;
      __bf16* l = &Vs[buf][row][ch];
#pragma unroll
      for (int c = 0; c < 4; ++c) async_cp16(g + c * 8, l + c * 8);
    }
  };

  v16bf aq[2];
#pragma unroll
  for (int kk = 0; kk < 2; ++kk)
    aq[kk] = load_frag(Qb + ((long)bh * T + qRow0) * Dh + kk * 32, Dh);

  const v8f zero = {0.f, 0.f, 0.f, 0.f, 0.f, 0.f, 0.f, 0.f};
  v8f o[4];
  float mrow[8], lrow[8];
#pragma unroll
  for (int dj = 0; dj < 4; ++dj) o[dj] = zero;
#pragma unroll
  for (int r = 0; r < 8; ++r) { mrow[r] = -3.0e38f; lrow[r] = 0.f; }

  issueKV(0, 0);
  for (int kb = 0; kb < T; kb += 128) {
    const int cur = (kb >> 7) & 1;
    wait_async0();
    __syncthreads();
    if (kb + 128 < T) issueKV(cur ^ 1, kb + 128);

    // S = Q K^T (16 queries x 128 keys), contraction over d=64
    v8f s[8];
#pragma unroll
    for (int nj = 0; nj < 8; ++nj) {
      v8f z = zero;
      z = wmma_bf16(aq[0], load_frag(&Ks[cur][nj * 16][0], 72), z);
      z = wmma_bf16(aq[1], load_frag(&Ks[cur][nj * 16][32], 72), z);
      s[nj] = z;
    }

    // scale by C^-0.5 = 1/32 ; per-row running max across 8 tiles + 16 lanes
    float mt[8];
#pragma unroll
    for (int r = 0; r < 8; ++r) mt[r] = -3.0e38f;
#pragma unroll
    for (int nj = 0; nj < 8; ++nj)
#pragma unroll
      for (int r = 0; r < 8; ++r) {
        const float v = s[nj][r] * 0.03125f;
        s[nj][r] = v;
        mt[r] = fmaxf(mt[r], v);
      }
#pragma unroll
    for (int r = 0; r < 8; ++r)
#pragma unroll
      for (int off = 8; off >= 1; off >>= 1)
        mt[r] = fmaxf(mt[r], __shfl_xor(mt[r], off, 32));

    float alpha[8];
#pragma unroll
    for (int r = 0; r < 8; ++r) {
      const float mn = fmaxf(mrow[r], mt[r]);
      alpha[r] = __builtin_amdgcn_exp2f((mrow[r] - mn) * LOG2E);
      mrow[r] = mn;
    }

    // P = exp(S - m): spill bf16 to per-wave LDS strip, accumulate row sums
    float rs[8];
#pragma unroll
    for (int r = 0; r < 8; ++r) rs[r] = 0.f;
#pragma unroll
    for (int nj = 0; nj < 8; ++nj)
#pragma unroll
      for (int r = 0; r < 8; ++r) {
        const float p = __builtin_amdgcn_exp2f((s[nj][r] - mrow[r]) * LOG2E);
        Ps[wv][r + 8 * half][nj * 16 + ln] = (__bf16)p;
        rs[r] += p;
      }
#pragma unroll
    for (int r = 0; r < 8; ++r) {
#pragma unroll
      for (int off = 8; off >= 1; off >>= 1) rs[r] += __shfl_xor(rs[r], off, 32);
      lrow[r] = lrow[r] * alpha[r] + rs[r];
    }
#pragma unroll
    for (int dj = 0; dj < 4; ++dj)
#pragma unroll
      for (int r = 0; r < 8; ++r) o[dj][r] *= alpha[r];

    // O += P V (contraction over 128 keys, 4 k-steps)
#pragma unroll
    for (int kk = 0; kk < 4; ++kk) {
      const v16bf ap = load_frag(&Ps[wv][0][kk * 32], 136);
#pragma unroll
      for (int dj = 0; dj < 4; ++dj)
        o[dj] = wmma_bf16(ap, load_frag(&Vs[cur][dj * 16][kk * 32], 136), o[dj]);
    }
  }

  // normalize + store AO[b, t, h*64+d] bf16 (feeds output projection)
#pragma unroll
  for (int dj = 0; dj < 4; ++dj)
#pragma unroll
    for (int r = 0; r < 8; ++r) {
      const float v = o[dj][r] / lrow[r];
      const int t = qRow0 + r + 8 * half;
      const int col = h * Dh + dj * 16 + ln;
      AO[((long)b * T + t) * 1024 + col] = (__bf16)v;
    }
}

// ---------------------------------------------------------------------------
// Kernel 3: output projection out = AO @ Wp^T (bf16 x bf16 -> f32).
// Same tiling/async double-buffering as qkv_proj.
// ---------------------------------------------------------------------------
__global__ __launch_bounds__(256) void out_proj(
    const __bf16* __restrict__ AO, const __bf16* __restrict__ Wpbf,
    float* __restrict__ out) {
  constexpr int Kdim = 1024;
  __shared__ __attribute__((aligned(16))) __bf16 As[2][256][40];
  __shared__ __attribute__((aligned(16))) __bf16 Bs[2][128][40];

  const int tid = threadIdx.x;
  const long aBase = (long)blockIdx.y * 256 * Kdim;
  const long bBase = (long)blockIdx.x * 128 * Kdim;
  const int wv = tid >> 5, waveM = wv >> 1, waveN = wv & 1;
  const int lane = tid & 31, half = lane >> 4, ln = lane & 15;

  auto issueAB = [&](int buf, int kt) {
    const __bf16* gA = AO + aBase + (long)tid * Kdim + kt;
    __bf16* lA = &As[buf][tid][0];
#pragma unroll
    for (int c = 0; c < 4; ++c) async_cp16(gA + c * 8, lA + c * 8);
    const int br = tid >> 1, bc = (tid & 1) * 16;
    const __bf16* gB = Wpbf + bBase + (long)br * Kdim + kt + bc;
    __bf16* lB = &Bs[buf][br][bc];
    async_cp16(gB, lB);
    async_cp16(gB + 8, lB + 8);
  };

  const v8f zero = {0.f, 0.f, 0.f, 0.f, 0.f, 0.f, 0.f, 0.f};
  v8f acc[4][4];
#pragma unroll
  for (int mi = 0; mi < 4; ++mi)
#pragma unroll
    for (int ni = 0; ni < 4; ++ni) acc[mi][ni] = zero;

  issueAB(0, 0);
  for (int kt = 0; kt < Kdim; kt += 32) {
    const int cur = (kt >> 5) & 1;
    wait_async0();
    __syncthreads();
    if (kt + 32 < Kdim) issueAB(cur ^ 1, kt + 32);

    v16bf a[4], b[4];
#pragma unroll
    for (int mi = 0; mi < 4; ++mi)
      a[mi] = load_frag(&As[cur][waveM * 64 + mi * 16][0], 40);
#pragma unroll
    for (int ni = 0; ni < 4; ++ni)
      b[ni] = load_frag(&Bs[cur][waveN * 64 + ni * 16][0], 40);
#pragma unroll
    for (int mi = 0; mi < 4; ++mi)
#pragma unroll
      for (int ni = 0; ni < 4; ++ni)
        acc[mi][ni] = wmma_bf16(a[mi], b[ni], acc[mi][ni]);
  }

#pragma unroll
  for (int mi = 0; mi < 4; ++mi)
#pragma unroll
    for (int ni = 0; ni < 4; ++ni)
#pragma unroll
      for (int r = 0; r < 8; ++r) {
        const int m = blockIdx.y * 256 + waveM * 64 + mi * 16 + r + 8 * half;
        const int n = blockIdx.x * 128 + waveN * 64 + ni * 16 + ln;
        out[(long)m * 1024 + n] = acc[mi][ni][r];
      }
}

// ---------------------------------------------------------------------------
extern "C" void kernel_launch(void* const* d_in, const int* in_sizes, int n_in,
                              void* d_out, int out_size, void* d_ws, size_t ws_size,
                              hipStream_t stream) {
  const float* Y  = (const float*)d_in[0];
  const float* X  = (const float*)d_in[1];
  const float* Wq = (const float*)d_in[2];
  const float* Wk = (const float*)d_in[3];
  const float* Wv = (const float*)d_in[4];
  const float* Wp = (const float*)d_in[5];
  float* out = (float*)d_out;

  constexpr int NACT = 2 * 2048 * 1024;  // 4,194,304
  constexpr int NW   = 1024 * 1024;      // 1,048,576

  char* ws = (char*)d_ws;
  size_t off = 0;
  auto alloc = [&](size_t bytes) { char* p = ws + off; off += bytes; return p; };
  __bf16* Ybf  = (__bf16*)alloc((size_t)NACT * 2);
  __bf16* Xbf  = (__bf16*)alloc((size_t)NACT * 2);
  __bf16* Wqbf = (__bf16*)alloc((size_t)NW * 2);
  __bf16* Wkbf = (__bf16*)alloc((size_t)NW * 2);
  __bf16* Wvbf = (__bf16*)alloc((size_t)NW * 2);
  __bf16* Wpbf = (__bf16*)alloc((size_t)NW * 2);
  __bf16* Qb   = (__bf16*)alloc((size_t)NACT * 2);
  __bf16* Kb   = (__bf16*)alloc((size_t)NACT * 2);
  __bf16* Vtb  = (__bf16*)alloc((size_t)NACT * 2);
  __bf16* AO   = (__bf16*)alloc((size_t)NACT * 2);

  // 0) one-pass f32 -> bf16 (L2-resident; frees hot loops from v_cvt VALU work)
  cvt_bf16<<<NACT / 2048, 256, 0, stream>>>(Y, Ybf, NACT);
  cvt_bf16<<<NACT / 2048, 256, 0, stream>>>(X, Xbf, NACT);
  cvt_bf16<<<NW / 2048, 256, 0, stream>>>(Wq, Wqbf, NW);
  cvt_bf16<<<NW / 2048, 256, 0, stream>>>(Wk, Wkbf, NW);
  cvt_bf16<<<NW / 2048, 256, 0, stream>>>(Wv, Wvbf, NW);
  cvt_bf16<<<NW / 2048, 256, 0, stream>>>(Wp, Wpbf, NW);

  // 1) Q/K/V projections: M tiles 4096/256=16, N tiles 1024/128=8, z selects
  qkv_proj<<<dim3(8, 16, 3), 256, 0, stream>>>(Ybf, Xbf, Wqbf, Wkbf, Wvbf,
                                               Qb, Kb, Vtb);
  // 2) flash attention: 16 query tiles x 16 heads x 2 batch
  attn<<<dim3(16, 16, 2), 256, 0, stream>>>(Qb, Kb, Vtb, AO);
  // 3) output projection
  out_proj<<<dim3(8, 16), 256, 0, stream>>>(AO, Wpbf, out);
}
